// GroupedQueryAttention_32985348833688
// MI455X (gfx1250) — matrix operand order repaired and checked by reading, not compiled, not run on hardware
//
#include <hip/hip_runtime.h>
#include <hip/hip_bf16.h>
#include <math.h>

#define B_    2
#define T_    2048
#define E_    2048
#define G_    4
#define QPG_  4
#define D_    128
#define H_    16          // G*QPG
#define NQ_   2048        // H*D
#define NKV_  1024        // 2*G*D
#define MTOT_ 4096        // B*T
#define KB_   128         // K-block staged in LDS per GEMM pipeline stage

typedef __bf16 v16bf __attribute__((ext_vector_type(16)));
typedef __bf16 v8bf  __attribute__((ext_vector_type(8)));
typedef float  v8f   __attribute__((ext_vector_type(8)));
typedef int    v4i   __attribute__((ext_vector_type(4)));

#define CAT16(lo, hi) __builtin_shufflevector((lo), (hi), 0,1,2,3,4,5,6,7,8,9,10,11,12,13,14,15)
#define WMMA_BF16(a, b, c) __builtin_amdgcn_wmma_f32_16x16x32_bf16(false, (a), false, (b), (short)0, (c), false, false)

#define NEG_INF (-1e30f)
#define SCALE_  (0.08838834764831845f)   // 1/sqrt(128)

// ---- CDNA5 async global->LDS copy (ASYNCcnt-tracked), with safe fallback ----
#if defined(__has_builtin)
#  if __has_builtin(__builtin_amdgcn_global_load_async_to_lds_b128)
#    define HAVE_ASYNC_LDS 1
#  endif
#endif

#ifdef HAVE_ASYNC_LDS
// signature (probe-confirmed by clang diagnostic): (v4i AS1* src, v4i AS3* dst, imm, imm)
#define ASYNC_COPY16(l, g)                                                  \
  __builtin_amdgcn_global_load_async_to_lds_b128(                           \
      (__attribute__((address_space(1))) v4i*)(g),                          \
      (__attribute__((address_space(3))) v4i*)(l), 0, 0)
#  if __has_builtin(__builtin_amdgcn_s_wait_asynccnt)
#    define WAIT_ASYNC(n) __builtin_amdgcn_s_wait_asynccnt(n)
#  else
#    define WAIT_ASYNC(n) asm volatile("s_wait_asynccnt %0" ::"i"(n) : "memory")
#  endif
#else
#define ASYNC_COPY16(l, g) (*(v8bf*)(l) = *(const v8bf*)(g))
#define WAIT_ASYNC(n) ((void)0)
#endif

__device__ __forceinline__ float pe_val(int t, int d) {
  int i = d & ~1;
  float inv = __expf(-9.210340371976184f * (float)i * (1.0f / 128.0f));
  float ang = (float)t * inv;
  return (d & 1) ? __cosf(ang) : __sinf(ang);
}

// ---------------- prep kernels ----------------
__global__ void cvt_f32_bf16(const float* __restrict__ in, __bf16* __restrict__ out, int n) {
  int i = blockIdx.x * blockDim.x + threadIdx.x;
  if (i < n) out[i] = (__bf16)in[i];
}

// in: (K,N) row-major f32 ; out: (N,K) row-major bf16
__global__ void transpose_cvt(const float* __restrict__ in, __bf16* __restrict__ out, int K, int N) {
  int i = blockIdx.x * blockDim.x + threadIdx.x;
  if (i >= K * N) return;
  int n = i / K, k = i - n * K;
  out[i] = (__bf16)in[(long)k * N + n];
}

// ---------------- staged GEMM mainloop (M=32 x N=256 per 128-thread block) ----------------
// The 32xKB_ A tile is double-buffered in LDS via ASYNCcnt-tracked copies;
// each B operand streams from global and is reused for two WMMAs (2 M-subtiles).
__device__ __forceinline__ void stage_A32(const __bf16* __restrict__ A, int K, int kb,
                                          __bf16* lds, int tid) {
#pragma unroll
  for (int p = 0; p < 4; ++p) {
    int c = p * 128 + tid;          // 512 x 16-byte chunks = 32 rows x 256 B
    int row = c >> 4;
    int cole = (c & 15) * 8;        // element offset within row
    const __bf16* g = A + (long)row * K + kb + cole;
    __bf16* l = lds + row * KB_ + cole;
    ASYNC_COPY16(l, g);
  }
}

__device__ __forceinline__ void wmma_mainloop32(const __bf16* __restrict__ A,   // A[m0*K]
                                                const __bf16* __restrict__ Bt,  // Bt[n0*K]
                                                int K, int tid, __bf16* ldsA, v8f acc[2][4]) {
  const int lane = tid & 31;
  const int half = lane >> 4;
  const int sub  = lane & 15;

  stage_A32(A, K, 0, ldsA, tid);
  int buf = 0;
  for (int kb = 0; kb < K; kb += KB_) {
    if (kb + KB_ < K) {
      stage_A32(A, K, kb + KB_, ldsA + (buf ^ 1) * (32 * KB_), tid);
      WAIT_ASYNC(4);                 // previous stage's 4 async ops done; next 4 in flight
    } else {
      WAIT_ASYNC(0);
    }
    __syncthreads();

    const __bf16* la = ldsA + buf * (32 * KB_) + (long)sub * KB_;
    const __bf16* bb = Bt + (long)sub * K + kb + half * 16;
#pragma unroll
    for (int kk = 0; kk < 4; ++kk) {
      v16bf b[4];
#pragma unroll
      for (int j = 0; j < 4; ++j)
        b[j] = *(const v16bf*)(bb + (long)j * 16 * K + kk * 32);
      v8bf lo0 = *(const v8bf*)(la + kk * 32 + half * 16);
      v8bf hi0 = *(const v8bf*)(la + kk * 32 + half * 16 + 8);
      v16bf a0 = CAT16(lo0, hi0);
      v8bf lo1 = *(const v8bf*)(la + 16 * KB_ + kk * 32 + half * 16);
      v8bf hi1 = *(const v8bf*)(la + 16 * KB_ + kk * 32 + half * 16 + 8);
      v16bf a1 = CAT16(lo1, hi1);
#pragma unroll
      for (int j = 0; j < 4; ++j) {
        acc[0][j] = WMMA_BF16(a0, b[j], acc[0][j]);
        acc[1][j] = WMMA_BF16(a1, b[j], acc[1][j]);
      }
    }
    __syncthreads();                 // protect LDS buffer reuse next-next stage
    buf ^= 1;
  }
}

// ---------------- Q projection ----------------
__global__ void __launch_bounds__(128) gemm_q_kernel(const __bf16* __restrict__ xb,
                                                     const __bf16* __restrict__ wqT,
                                                     const float* __restrict__ bq,
                                                     __bf16* __restrict__ qbf) {
  __shared__ __align__(16) __bf16 ldsA[2 * 32 * KB_];
  const int tid = threadIdx.x, wave = tid >> 5, lane = tid & 31;
  const int half = lane >> 4, sub = lane & 15;
  const int m0 = blockIdx.x * 32;
  const int n0 = blockIdx.y * 256 + wave * 64;
  v8f acc[2][4] = {};
  wmma_mainloop32(xb + (long)m0 * E_, wqT + (long)n0 * E_, E_, tid, ldsA, acc);
#pragma unroll
  for (int s = 0; s < 2; ++s) {
#pragma unroll
    for (int j = 0; j < 4; ++j) {
      int n = n0 + j * 16 + sub;
      int h = n >> 7, d = n & 127;
      float bias = bq[n];
#pragma unroll
      for (int r = 0; r < 8; ++r) {
        int gm = m0 + s * 16 + r + half * 8;
        int b = gm >> 11, t = gm & (T_ - 1);
        float v = (acc[s][j][r] + bias + pe_val(t, d)) * SCALE_;
        qbf[(((long)b * H_ + h) * T_ + t) * D_ + d] = (__bf16)v;
      }
    }
  }
}

// ---------------- KV projection: k -> (b,g,t,d)+PE ; v -> (b,g,d,t) ----------------
__global__ void __launch_bounds__(128) gemm_kv_kernel(const __bf16* __restrict__ xb,
                                                      const __bf16* __restrict__ wkvT,
                                                      const float* __restrict__ bkv,
                                                      __bf16* __restrict__ kbf,
                                                      __bf16* __restrict__ vTb) {
  __shared__ __align__(16) __bf16 ldsA[2 * 32 * KB_];
  const int tid = threadIdx.x, wave = tid >> 5, lane = tid & 31;
  const int half = lane >> 4, sub = lane & 15;
  const int m0 = blockIdx.x * 32;
  const int n0 = blockIdx.y * 256 + wave * 64;
  v8f acc[2][4] = {};
  wmma_mainloop32(xb + (long)m0 * E_, wkvT + (long)n0 * E_, E_, tid, ldsA, acc);
#pragma unroll
  for (int s = 0; s < 2; ++s) {
#pragma unroll
    for (int j = 0; j < 4; ++j) {
      int n = n0 + j * 16 + sub;
      int g = n >> 8, r2 = n & 255;
      float bias = bkv[n];
#pragma unroll
      for (int r = 0; r < 8; ++r) {
        int gm = m0 + s * 16 + r + half * 8;
        int b = gm >> 11, t = gm & (T_ - 1);
        float v = acc[s][j][r] + bias;
        if (r2 < 128) {
          int d = r2;
          kbf[(((long)b * G_ + g) * T_ + t) * D_ + d] = (__bf16)(v + pe_val(t, d));
        } else {
          int d = r2 - 128;
          vTb[(((long)b * G_ + g) * D_ + d) * T_ + t] = (__bf16)v;
        }
      }
    }
  }
}

// ---------------- causal flash attention: one wave per 16 query rows ----------------
__global__ void __launch_bounds__(128) attn_kernel(const __bf16* __restrict__ qbf,
                                                   const __bf16* __restrict__ kbf,
                                                   const __bf16* __restrict__ vTb,
                                                   __bf16* __restrict__ attn) {
  __shared__ __align__(16) __bf16 Pbuf[4][16][32];
  const int wave = threadIdx.x >> 5, lane = threadIdx.x & 31;
  const int half = lane >> 4, sub = lane & 15;
  const int bh = blockIdx.y;
  const int b = bh >> 4, h = bh & 15, g = h >> 2;
  const int qbase = (blockIdx.x * 4 + wave) * 16;

  const __bf16* qrow = qbf + (((long)b * H_ + h) * T_ + qbase) * D_ + (long)sub * D_ + half * 16;
  v16bf qa[4];
#pragma unroll
  for (int kk = 0; kk < 4; ++kk) {
    v8bf lo = *(const v8bf*)(qrow + kk * 32);
    v8bf hi = *(const v8bf*)(qrow + kk * 32 + 8);
    qa[kk] = CAT16(lo, hi);
  }

  v8f o[8] = {};
  float rmax[8], rsum[8];
#pragma unroll
  for (int r = 0; r < 8; ++r) { rmax[r] = NEG_INF; rsum[r] = 0.0f; }

  const __bf16* kb_ptr = kbf + ((long)b * G_ + g) * T_ * D_;
  const __bf16* vb_ptr = vTb + ((long)b * G_ + g) * D_ * T_;
  const int qend = qbase + 15;

  for (int kb = 0; kb <= qend; kb += 32) {
    // ---- S = Q Kt over kv cols [kb, kb+32) ----
    v8f c0 = {}, c1 = {};
#pragma unroll
    for (int kk = 0; kk < 4; ++kk) {
      const __bf16* b0 = kb_ptr + (long)(kb + sub) * D_ + kk * 32 + half * 16;
      const __bf16* b1 = kb_ptr + (long)(kb + 16 + sub) * D_ + kk * 32 + half * 16;
      v16bf vb0 = *(const v16bf*)b0;
      v16bf vb1 = *(const v16bf*)b1;
      c0 = WMMA_BF16(qa[kk], vb0, c0);
      c1 = WMMA_BF16(qa[kk], vb1, c1);
    }
    // ---- issue V loads now; softmax VALU below hides their latency ----
    v16bf vv[8];
#pragma unroll
    for (int j = 0; j < 8; ++j)
      vv[j] = *(const v16bf*)(vb_ptr + (long)(j * 16 + sub) * T_ + kb + half * 16);

    // ---- causal mask + online softmax (row m = r + 8*half in one 16-lane half) ----
    const int col0 = kb + sub, col1 = col0 + 16;
#pragma unroll
    for (int r = 0; r < 8; ++r) {
      int m = r + half * 8;
      int row = qbase + m;
      float s0 = (col0 <= row) ? c0[r] : NEG_INF;
      float s1 = (col1 <= row) ? c1[r] : NEG_INF;
      float mx = fmaxf(s0, s1);
      mx = fmaxf(mx, __shfl_xor(mx, 1, 32));
      mx = fmaxf(mx, __shfl_xor(mx, 2, 32));
      mx = fmaxf(mx, __shfl_xor(mx, 4, 32));
      mx = fmaxf(mx, __shfl_xor(mx, 8, 32));
      float newm = fmaxf(rmax[r], mx);
      float corr = __expf(rmax[r] - newm);
      float p0 = __expf(s0 - newm);
      float p1 = __expf(s1 - newm);
      float ps = p0 + p1;
      ps += __shfl_xor(ps, 1, 32);
      ps += __shfl_xor(ps, 2, 32);
      ps += __shfl_xor(ps, 4, 32);
      ps += __shfl_xor(ps, 8, 32);
      rsum[r] = rsum[r] * corr + ps;
      rmax[r] = newm;
#pragma unroll
      for (int j = 0; j < 8; ++j) o[j][r] *= corr;
      Pbuf[wave][m][sub]      = (__bf16)p0;
      Pbuf[wave][m][sub + 16] = (__bf16)p1;
    }
    // ---- reload P in A-operand layout ----
    v8bf plo = *(const v8bf*)&Pbuf[wave][sub][half * 16];
    v8bf phi = *(const v8bf*)&Pbuf[wave][sub][half * 16 + 8];
    v16bf pa = CAT16(plo, phi);
    // ---- O += P V ----
#pragma unroll
    for (int j = 0; j < 8; ++j) o[j] = WMMA_BF16(pa, vv[j], o[j]);
  }

  float rinv[8];
#pragma unroll
  for (int r = 0; r < 8; ++r) rinv[r] = 1.0f / rsum[r];
#pragma unroll
  for (int j = 0; j < 8; ++j) {
    int dcol = j * 16 + sub;
#pragma unroll
    for (int r = 0; r < 8; ++r) {
      int t = qbase + r + half * 8;
      attn[((long)b * T_ + t) * NQ_ + h * D_ + dcol] = (__bf16)(o[j][r] * rinv[r]);
    }
  }
}

// ---------------- output projection: out = attn @ wo (f32) ----------------
__global__ void __launch_bounds__(128) gemm_o_kernel(const __bf16* __restrict__ attn,
                                                     const __bf16* __restrict__ woT,
                                                     float* __restrict__ out) {
  __shared__ __align__(16) __bf16 ldsA[2 * 32 * KB_];
  const int tid = threadIdx.x, wave = tid >> 5, lane = tid & 31;
  const int half = lane >> 4, sub = lane & 15;
  const int m0 = blockIdx.x * 32;
  const int n0 = blockIdx.y * 256 + wave * 64;
  v8f acc[2][4] = {};
  wmma_mainloop32(attn + (long)m0 * NQ_, woT + (long)n0 * NQ_, NQ_, tid, ldsA, acc);
#pragma unroll
  for (int s = 0; s < 2; ++s) {
#pragma unroll
    for (int j = 0; j < 4; ++j) {
      int n = n0 + j * 16 + sub;
#pragma unroll
      for (int r = 0; r < 8; ++r) {
        int gm = m0 + s * 16 + r + half * 8;
        out[(long)gm * E_ + n] = acc[s][j][r];
      }
    }
  }
}

extern "C" void kernel_launch(void* const* d_in, const int* in_sizes, int n_in,
                              void* d_out, int out_size, void* d_ws, size_t ws_size,
                              hipStream_t stream) {
  const float* x   = (const float*)d_in[0];
  const float* wq  = (const float*)d_in[1];
  const float* bq  = (const float*)d_in[2];
  const float* wkv = (const float*)d_in[3];
  const float* bkv = (const float*)d_in[4];
  const float* wo  = (const float*)d_in[5];
  float* out = (float*)d_out;

  char* ws = (char*)d_ws;
  size_t off = 0;
  __bf16* xb   = (__bf16*)(ws + off); off += (size_t)MTOT_ * E_ * 2;
  __bf16* wqT  = (__bf16*)(ws + off); off += (size_t)NQ_   * E_ * 2;
  __bf16* wkvT = (__bf16*)(ws + off); off += (size_t)NKV_  * E_ * 2;
  __bf16* woT  = (__bf16*)(ws + off); off += (size_t)E_    * NQ_ * 2;
  __bf16* qbf  = (__bf16*)(ws + off); off += (size_t)B_ * H_ * T_ * D_ * 2;
  __bf16* kbf  = (__bf16*)(ws + off); off += (size_t)B_ * G_ * T_ * D_ * 2;
  __bf16* vTb  = (__bf16*)(ws + off); off += (size_t)B_ * G_ * D_ * T_ * 2;
  __bf16* attn = (__bf16*)(ws + off);

  cvt_f32_bf16<<<(MTOT_ * E_ + 255) / 256, 256, 0, stream>>>(x, xb, MTOT_ * E_);
  transpose_cvt<<<(E_ * NQ_  + 255) / 256, 256, 0, stream>>>(wq,  wqT,  E_,  NQ_);
  transpose_cvt<<<(E_ * NKV_ + 255) / 256, 256, 0, stream>>>(wkv, wkvT, E_,  NKV_);
  transpose_cvt<<<(NQ_ * E_  + 255) / 256, 256, 0, stream>>>(wo,  woT,  NQ_, E_);

  gemm_q_kernel <<<dim3(MTOT_ / 32, NQ_  / 256), 128, 0, stream>>>(xb, wqT,  bq,  qbf);
  gemm_kv_kernel<<<dim3(MTOT_ / 32, NKV_ / 256), 128, 0, stream>>>(xb, wkvT, bkv, kbf, vTb);
  attn_kernel   <<<dim3(T_ / 64, B_ * H_),       128, 0, stream>>>(qbf, kbf, vTb, attn);
  gemm_o_kernel <<<dim3(MTOT_ / 32, E_  / 256),  128, 0, stream>>>(attn, woT, out);
}
